// PhyEncoder_38895223832829
// MI455X (gfx1250) — matrix operand order repaired and checked by reading, not compiled, not run on hardware
//
#include <hip/hip_runtime.h>
#include <hip/hip_bf16.h>
#include <math.h>

// ---------------- types for WMMA ----------------
typedef __attribute__((ext_vector_type(16))) __bf16 v16bf;
typedef __attribute__((ext_vector_type(8)))  float  v8f;

#define DEV __device__ __forceinline__

static constexpr int Bn = 16, Tn = 48, Vn = 1024, NQ = 46;
static constexpr int T1 = 46, T2 = 42;          // sconv T per block
static constexpr int R1 = Bn * 32 * T1;         // 23552 GEMM rows
static constexpr int R2 = Bn * 32 * T2;         // 21504

DEV unsigned short f2bf(float f) {              // f32 -> bf16 RNE
  unsigned u = __float_as_uint(f);
  unsigned r = u + 0x7FFFu + ((u >> 16) & 1u);
  return (unsigned short)(r >> 16);
}
DEV float sigmf(float x) { return 1.0f / (1.0f + __expf(-x)); }

DEV float blk_reduce256(float v, float* sred) { // blockDim.x == 256
  int tid = threadIdx.x;
  sred[tid] = v; __syncthreads();
  for (int st = 128; st > 0; st >>= 1) {
    if (tid < st) sred[tid] += sred[tid + st];
    __syncthreads();
  }
  float r = sred[0]; __syncthreads();
  return r;
}

// ---------------- graph prep ----------------
__global__ __launch_bounds__(256) void k_degree(const float* __restrict__ g,
                                                float* __restrict__ dinv) {
  __shared__ float sred[256];
  int n = blockIdx.x;
  float s = 0.f;
  for (int m = threadIdx.x; m < Vn; m += 256) s += g[(size_t)n * Vn + m];
  float tot = blk_reduce256(s, sred);
  if (threadIdx.x == 0) dinv[n] = rsqrtf(tot + 1.0f);   // +1 self loop
}

__global__ __launch_bounds__(256) void k_laplacian(const float* __restrict__ g,
                                                   const float* __restrict__ dinv,
                                                   float* __restrict__ L) {
  size_t idx = (size_t)blockIdx.x * 256 + threadIdx.x;
  int n = (int)(idx >> 10), m = (int)(idx & (Vn - 1));
  float eye = (n == m) ? 1.f : 0.f;
  L[idx] = eye - dinv[n] * (g[idx] + eye) * dinv[m];
}

__global__ __launch_bounds__(256) void k_cvtL(const float* __restrict__ L,
                                              unsigned short* __restrict__ A,
                                              unsigned short* __restrict__ Bt) {
  size_t idx = (size_t)blockIdx.x * 256 + threadIdx.x;
  int i = (int)(idx >> 10), p = (int)(idx & (Vn - 1));
  A[idx]  = f2bf(L[idx]);
  Bt[idx] = f2bf(L[(size_t)p * Vn + i]);   // transpose
}

// LkT[n][k][m] = Lk[k][n][m]  (bf16, N-major for the NT GEMM operand)
__global__ __launch_bounds__(256) void k_buildLkT(const float* __restrict__ L,
                                                  const float* __restrict__ LL,
                                                  unsigned short* __restrict__ LkT) {
  size_t idx = (size_t)blockIdx.x * 256 + threadIdx.x;
  int n = (int)(idx >> 10), m = (int)(idx & (Vn - 1));
  float eye = (n == m) ? 1.f : 0.f;
  size_t base = ((size_t)n * 3) * Vn + m;
  LkT[base]           = f2bf(eye);
  LkT[base + Vn]      = f2bf(L[idx]);
  LkT[base + 2 * Vn]  = f2bf(2.f * LL[idx] - eye);
}

// ---------------- WMMA bf16 GEMM: C[M,N] = A[M,K] * Bt[N,K]^T ----------------
// 256 threads = 8 waves; block tile 64x128, K-step 32; each wave: 16x64 of C
// (one A fragment feeds 4 WMMAs). Staging uses GLOBAL_LOAD_ASYNC_TO_LDS_B128
// (ASYNCcnt) with double-buffered LDS: next K-tile is in flight while the
// current one is consumed by the matrix pipe.
static constexpr int LDS_A  = 64 * 40;            // ushorts, stride-40 padded
static constexpr int LDS_B  = 128 * 40;
static constexpr int LDS_BUF = LDS_A + LDS_B;     // 7680 ushorts = 15360 B

__global__ __launch_bounds__(256) void k_gemm_nt(
    const unsigned short* __restrict__ A, const unsigned short* __restrict__ Bt,
    float* __restrict__ C, const float* __restrict__ res, const float* __restrict__ bias,
    int M, int N, int K, int bias_div, int bias_mod, int do_relu) {
  __shared__ __align__(16) unsigned short smem[2 * LDS_BUF];
  const int tid = threadIdx.x;
  const int lane = tid & 31, wid = tid >> 5;
  const int mi = wid & 3;        // M sub-tile 0..3
  const int nq = wid >> 2;       // N half 0..1 (64 cols each)
  const size_t rBase = (size_t)blockIdx.y * 64;
  const size_t nBase = (size_t)blockIdx.x * 128;

  v8f acc[4];
#pragma unroll
  for (int i = 0; i < 4; ++i) acc[i] = (v8f){0.f,0.f,0.f,0.f,0.f,0.f,0.f,0.f};

  // staging: thread covers one 16B chunk of the A tile and two of the B tile
  const int lr = tid >> 2, lc = (tid & 3) * 8;
  const unsigned short* Ag  = A  + (rBase + lr) * (size_t)K + lc;
  const unsigned short* Bg0 = Bt + (nBase + lr) * (size_t)K + lc;
  const unsigned short* Bg1 = Bt + (nBase + 64 + lr) * (size_t)K + lc;
  const unsigned ldsBase = (unsigned)(size_t)smem;           // LDS byte offset
  const unsigned sA  = ldsBase + (unsigned)(lr * 40 + lc) * 2;
  const unsigned sB0 = ldsBase + (unsigned)(LDS_A + lr * 40 + lc) * 2;
  const unsigned sB1 = sB0 + 64 * 40 * 2;

  // fragment geometry (ISA 7.12.2 layouts)
  const int fr = lane & 15, fh = lane >> 4;

  auto issueTile = [&](int bufi, int k0) {
    unsigned off = bufi ? (unsigned)(LDS_BUF * 2) : 0u;
    unsigned long long ga  = (unsigned long long)(size_t)(Ag  + k0);
    unsigned long long gb0 = (unsigned long long)(size_t)(Bg0 + k0);
    unsigned long long gb1 = (unsigned long long)(size_t)(Bg1 + k0);
    asm volatile("global_load_async_to_lds_b128 %0, %1, off"
                 :: "v"(sA + off),  "v"(ga)  : "memory");
    asm volatile("global_load_async_to_lds_b128 %0, %1, off"
                 :: "v"(sB0 + off), "v"(gb0) : "memory");
    asm volatile("global_load_async_to_lds_b128 %0, %1, off"
                 :: "v"(sB1 + off), "v"(gb1) : "memory");
  };

  const int nsteps = K >> 5;
  issueTile(0, 0);
  for (int it = 0; it < nsteps; ++it) {
    if (it + 1 < nsteps) {
      issueTile((it + 1) & 1, (it + 1) * 32);      // prefetch next tile
      asm volatile("s_wait_asynccnt 3" ::: "memory");  // oldest tile landed
    } else {
      asm volatile("s_wait_asynccnt 0" ::: "memory");
    }
    __syncthreads();
    const unsigned short* sb = smem + (it & 1) * LDS_BUF;
    union { v16bf v; uint4 q[2]; } af;
    const unsigned short* ap = sb + (mi * 16 + fr) * 40 + fh * 8;
    af.q[0] = *(const uint4*)ap;  af.q[1] = *(const uint4*)(ap + 16);
#pragma unroll
    for (int ni = 0; ni < 4; ++ni) {
      union { v16bf v; uint4 q[2]; } bf_;
      const unsigned short* bp =
          sb + LDS_A + ((nq * 64 + ni * 16 + fr) * 40 + fh * 8);
      bf_.q[0] = *(const uint4*)bp;  bf_.q[1] = *(const uint4*)(bp + 16);
      acc[ni] = __builtin_amdgcn_wmma_f32_16x16x32_bf16(
          false, af.v, false, bf_.v, (short)0, acc[ni], false, false);
    }
    __syncthreads();
  }

  // C frag: VGPR j -> row j + 8*fh ; col = lane&15
#pragma unroll
  for (int ni = 0; ni < 4; ++ni) {
    const size_t n0 = nBase + nq * 64 + ni * 16 + fr;
#pragma unroll
    for (int j = 0; j < 8; ++j) {
      size_t r = rBase + mi * 16 + j + 8 * fh;
      float v0 = acc[ni][j];
      if (bias) v0 += bias[(int)((r / bias_div) % bias_mod)];
      if (res)  v0 += res[r * (size_t)N + n0];
      if (do_relu) v0 = fmaxf(v0, 0.f);
      C[r * (size_t)N + n0] = v0;
    }
  }
}

// ---------------- block 1: temporal GLU conv (pads input, builds diff channel) --
__global__ __launch_bounds__(256) void k_tconv1(const float* __restrict__ x0,
                                                const float* __restrict__ w11,
                                                const float* __restrict__ b11,
                                                float* __restrict__ out) {
  int v = blockIdx.x * 256 + threadIdx.x;
  int t = blockIdx.y;            // 0..47
  int b = blockIdx.z;
  float in[3][3];
#pragma unroll
  for (int dt = 0; dt < 3; ++dt) {
    int ts = t + dt - 2;         // padded row -> real x0 row
    float a0 = 0.f, a1 = 0.f;
    if (ts >= 0) {
      size_t o_ = (((size_t)b * Tn + ts) * Vn + v) * 2;
      a0 = x0[o_]; a1 = x0[o_ + 1];
    }
    in[0][dt] = a0; in[1][dt] = a1; in[2][dt] = a0 - a1;
  }
  size_t ob = (((size_t)b * 32) * Tn + t) * Vn + v;
#pragma unroll
  for (int o = 0; o < 32; ++o) {
    float a = b11[o], g = b11[o + 32];
#pragma unroll
    for (int c = 0; c < 3; ++c)
#pragma unroll
      for (int dt = 0; dt < 3; ++dt) {
        a += w11[(o * 3 + c) * 3 + dt] * in[c][dt];
        g += w11[((o + 32) * 3 + c) * 3 + dt] * in[c][dt];
      }
    float xin = (o < 3) ? in[o][2] : 0.f;    // channel pad 3->32, slice [2:]
    out[ob + (size_t)o * Tn * Vn] = (a + xin) * sigmf(g);
  }
}

// ---------------- attention pooler (per (b,v) slice in LDS) ----------------
__global__ __launch_bounds__(64) void k_pooler(const float* __restrict__ X,
                                               const float* __restrict__ aw,
                                               const float* __restrict__ ab,
                                               float* __restrict__ out) {
  __shared__ float xs[32 * 48];
  int v = blockIdx.x, b = blockIdx.y;
  for (int e = threadIdx.x; e < 32 * 48; e += 64) {
    int c = e / 48, l = e % 48;
    xs[e] = X[(((size_t)b * 32 + c) * Tn + l) * Vn + v];
  }
  __syncthreads();
  int q = threadIdx.x;
  if (q >= NQ) return;
  float lg[48];
  float mx = -1e30f;
#pragma unroll
  for (int l = 0; l < 48; ++l) {
    float s = ab[q];
    for (int c = 0; c < 32; ++c) s += aw[q * 32 + c] * xs[c * 48 + l];
    lg[l] = s; mx = fmaxf(mx, s);
  }
  float den = 0.f;
#pragma unroll
  for (int l = 0; l < 48; ++l) { lg[l] = __expf(lg[l] - mx); den += lg[l]; }
  float inv = 1.f / den;
  for (int c = 0; c < 32; ++c) {
    float acc = 0.f;
#pragma unroll
    for (int l = 0; l < 48; ++l) acc += xs[c * 48 + l] * lg[l];
    acc = acc * inv + xs[c * 48 + (q + 2)];       // + x_in (rows -46:)
    out[(((size_t)b * 32 + c) * NQ + q) * Vn + v] = fmaxf(acc, 0.f);
  }
}

// ---------------- theta pre-mix: A[r, k*V+m] = sum_i th[i,o,k]*X[b,i,t,m] -------
__global__ __launch_bounds__(256) void k_mix(const float* __restrict__ X,
                                             const float* __restrict__ th,
                                             unsigned short* __restrict__ A,
                                             int T, int R) {
  size_t idx = (size_t)blockIdx.x * 256 + threadIdx.x;
  if (idx >= (size_t)R * Vn) return;
  int m = (int)(idx & (Vn - 1));
  int r = (int)(idx >> 10);
  int t = r % T; int o = (r / T) & 31; int b = r / (T * 32);
  const float* xp = X + (((size_t)b * 32) * T + t) * Vn + m;
  float a0 = 0.f, a1 = 0.f, a2 = 0.f;
  for (int i = 0; i < 32; ++i) {
    float xv = xp[(size_t)i * T * Vn];
    a0 += th[(i * 32 + o) * 3 + 0] * xv;
    a1 += th[(i * 32 + o) * 3 + 1] * xv;
    a2 += th[(i * 32 + o) * 3 + 2] * xv;
  }
  size_t ab = (size_t)r * 3072;
  A[ab + m]        = f2bf(a0);
  A[ab + 1024 + m] = f2bf(a1);
  A[ab + 2048 + m] = f2bf(a2);
}

// ---------------- temporal conv + residual + ReLU (32 -> 64 ch) ----------------
__global__ __launch_bounds__(256) void k_tconv_relu(const float* __restrict__ X,
                                                    const float* __restrict__ w,
                                                    const float* __restrict__ bias,
                                                    float* __restrict__ out,
                                                    int Tin, int Tout) {
  int v = blockIdx.x * 256 + threadIdx.x;
  int t = blockIdx.y, b = blockIdx.z;
  float acc[64];
#pragma unroll
  for (int o = 0; o < 64; ++o) acc[o] = bias[o];
  for (int i = 0; i < 32; ++i)
#pragma unroll
    for (int dt = 0; dt < 3; ++dt) {
      float xv = X[(((size_t)b * 32 + i) * Tin + t + dt) * Vn + v];
#pragma unroll
      for (int o = 0; o < 64; ++o) acc[o] += w[(o * 32 + i) * 3 + dt] * xv;
    }
#pragma unroll
  for (int o = 0; o < 64; ++o) {
    float r = acc[o];
    if (o < 32) r += X[(((size_t)b * 32 + o) * Tin + t + 2) * Vn + v];
    out[(((size_t)b * 64 + o) * Tout + t) * Vn + v] = fmaxf(r, 0.f);
  }
}

// ---------------- LayerNorm over (V,C) per (b,t), in place -----------------
__global__ __launch_bounds__(256) void k_ln(float* __restrict__ X,
                                            const float* __restrict__ g,
                                            const float* __restrict__ be,
                                            int C, int T) {
  __shared__ float sred[256];
  int t = blockIdx.x, b = blockIdx.y;
  const int N = C * Vn;
  float s = 0.f, ss = 0.f;
  for (int e = threadIdx.x; e < N; e += 256) {
    int c = e >> 10, v = e & (Vn - 1);
    float x = X[(((size_t)b * C + c) * T + t) * Vn + v];
    s += x; ss += x * x;
  }
  float mean = blk_reduce256(s, sred) / N;
  float var  = blk_reduce256(ss, sred) / N - mean * mean;
  float rstd = rsqrtf(var + 1e-5f);
  for (int e = threadIdx.x; e < N; e += 256) {
    int c = e >> 10, v = e & (Vn - 1);
    size_t ix = (((size_t)b * C + c) * T + t) * Vn + v;
    X[ix] = (X[ix] - mean) * rstd * g[v * C + c] + be[v * C + c];
  }
}

// ---------------- block 2 GLU conv with 1x1 align conv (64 -> 32 ch) -----------
__global__ __launch_bounds__(256) void k_tconv_glu2(const float* __restrict__ X,
                                                    const float* __restrict__ w,
                                                    const float* __restrict__ bias,
                                                    const float* __restrict__ aw,
                                                    const float* __restrict__ ab,
                                                    float* __restrict__ out,
                                                    int Tin, int Tout) {
  int v = blockIdx.x * 256 + threadIdx.x;
  int t = blockIdx.y;
  int bz = blockIdx.z; int b = bz >> 1, og = (bz & 1) * 16;
  float a1[16], ag[16], ain[16];
#pragma unroll
  for (int oo = 0; oo < 16; ++oo) {
    a1[oo] = bias[og + oo]; ag[oo] = bias[og + oo + 32]; ain[oo] = ab[og + oo];
  }
  for (int i = 0; i < 64; ++i) {
    float xv0 = X[(((size_t)b * 64 + i) * Tin + t)     * Vn + v];
    float xv1 = X[(((size_t)b * 64 + i) * Tin + t + 1) * Vn + v];
    float xv2 = X[(((size_t)b * 64 + i) * Tin + t + 2) * Vn + v];
#pragma unroll
    for (int oo = 0; oo < 16; ++oo) {
      int o = og + oo;
      a1[oo] += w[(o * 64 + i) * 3 + 0] * xv0 + w[(o * 64 + i) * 3 + 1] * xv1 +
                w[(o * 64 + i) * 3 + 2] * xv2;
      ag[oo] += w[((o + 32) * 64 + i) * 3 + 0] * xv0 +
                w[((o + 32) * 64 + i) * 3 + 1] * xv1 +
                w[((o + 32) * 64 + i) * 3 + 2] * xv2;
      ain[oo] += aw[o * 64 + i] * xv2;
    }
  }
#pragma unroll
  for (int oo = 0; oo < 16; ++oo) {
    int o = og + oo;
    out[(((size_t)b * 32 + o) * Tout + t) * Vn + v] =
        (a1[oo] + ain[oo]) * sigmf(ag[oo]);
  }
}

// ---------------- output temporal conv (kt = 40, 64 -> 128 ch) ------------------
__global__ __launch_bounds__(256) void k_tconv_out(const float* __restrict__ X,
                                                   const float* __restrict__ w,
                                                   const float* __restrict__ bias,
                                                   float* __restrict__ xc) {
  int v = blockIdx.x * 256 + threadIdx.x;
  int og = blockIdx.y * 16;   // 8 groups of 16 outputs
  int b = blockIdx.z;
  float acc[16];
#pragma unroll
  for (int oo = 0; oo < 16; ++oo) acc[oo] = bias[og + oo];
  for (int i = 0; i < 64; ++i)
    for (int dt = 0; dt < 40; ++dt) {
      float xv = X[(((size_t)b * 64 + i) * 40 + dt) * Vn + v];
#pragma unroll
      for (int oo = 0; oo < 16; ++oo)
        acc[oo] += w[((og + oo) * 64 + i) * 40 + dt] * xv;
    }
#pragma unroll
  for (int oo = 0; oo < 16; ++oo)
    xc[((size_t)b * 128 + og + oo) * Vn + v] = acc[oo];
}

__global__ __launch_bounds__(256) void k_glu_out(const float* __restrict__ xc,
                                                 const float* __restrict__ X,
                                                 float* __restrict__ zg) {
  int v = blockIdx.x * 256 + threadIdx.x;
  int c = blockIdx.y, b = blockIdx.z;
  float a = xc[((size_t)b * 128 + c) * Vn + v] +
            X[(((size_t)b * 64 + c) * 40 + 39) * Vn + v];   // x_in = x[:,:,39:,:]
  float g = xc[((size_t)b * 128 + 64 + c) * Vn + v];
  zg[((size_t)b * 64 + c) * Vn + v] = a * sigmf(g);
}

// ---------------- LN3: (B,64,1,V) -> normalized, transposed to (B,V,64) ---------
__global__ __launch_bounds__(256) void k_ln3(const float* __restrict__ zg,
                                             const float* __restrict__ g,
                                             const float* __restrict__ be,
                                             float* __restrict__ z) {
  __shared__ float sred[256];
  int b = blockIdx.x;
  const int N = 64 * Vn;
  float s = 0.f, ss = 0.f;
  for (int e = threadIdx.x; e < N; e += 256) {
    float x = zg[(size_t)b * N + e]; s += x; ss += x * x;
  }
  float mean = blk_reduce256(s, sred) / N;
  float var  = blk_reduce256(ss, sred) / N - mean * mean;
  float rstd = rsqrtf(var + 1e-5f);
  for (int e = threadIdx.x; e < N; e += 256) {
    int c = e >> 10, v = e & (Vn - 1);
    float x = zg[(size_t)b * N + e];
    z[(size_t)b * N + v * 64 + c] = (x - mean) * rstd * g[v * 64 + c] + be[v * 64 + c];
  }
}

// ---------------- residual graph conv on f_i / f_o ------------------------------
__global__ __launch_bounds__(256) void k_res(const float* __restrict__ x0,
                                             const float* __restrict__ L,
                                             const float* __restrict__ LL,
                                             const float* __restrict__ th,
                                             const float* __restrict__ bb,
                                             int ch, float* __restrict__ out) {
  __shared__ float sred[256];
  int n = blockIdx.x, b = blockIdx.y;
  size_t fbase = (((size_t)b * Tn + 47) * Vn) * 2 + ch;  // f[:, -1, :, ch]
  float s1 = 0.f, s2 = 0.f;
  for (int m = threadIdx.x; m < Vn; m += 256) {
    float fv = x0[fbase + (size_t)m * 2];
    s1 += L[(size_t)n * Vn + m] * fv;
    s2 += LL[(size_t)n * Vn + m] * fv;
  }
  float y1 = blk_reduce256(s1, sred);
  float u2 = blk_reduce256(s2, sred);
  float y0 = x0[fbase + (size_t)n * 2];
  float y2 = 2.f * u2 - y0;
  if (threadIdx.x < 64) {
    int o = threadIdx.x;
    float r = th[o * 3 + 0] * y0 + th[o * 3 + 1] * y1 + th[o * 3 + 2] * y2 +
              bb[o] + (o == 0 ? y0 : 0.f);   // channel-pad align residual
    out[((size_t)b * Vn + n) * 64 + o] = fmaxf(r, 0.f);
  }
}

// ---------------- (z + ri - ro) @ wlr^T + blr, ReLU -----------------------------
__global__ __launch_bounds__(64) void k_linear(const float* __restrict__ z,
                                               const float* __restrict__ ri,
                                               const float* __restrict__ ro,
                                               const float* __restrict__ wlr,
                                               const float* __restrict__ blr,
                                               float* __restrict__ zl) {
  __shared__ float s[64];
  int v = blockIdx.x, b = blockIdx.y;
  size_t base = ((size_t)b * Vn + v) * 64;
  int j = threadIdx.x;
  s[j] = z[base + j] + ri[base + j] - ro[base + j];
  __syncthreads();
  float acc = blr[j];
#pragma unroll
  for (int c = 0; c < 64; ++c) acc += s[c] * wlr[j * 64 + c];
  zl[base + j] = fmaxf(acc, 0.f);
}

// ---------------- final LN over (V,64) per b -> d_out ---------------------------
__global__ __launch_bounds__(256) void k_ln4(const float* __restrict__ zl,
                                             const float* __restrict__ g,
                                             const float* __restrict__ be,
                                             float* __restrict__ out) {
  __shared__ float sred[256];
  int b = blockIdx.x;
  const int N = 64 * Vn;
  float s = 0.f, ss = 0.f;
  for (int e = threadIdx.x; e < N; e += 256) {
    float x = zl[(size_t)b * N + e]; s += x; ss += x * x;
  }
  float mean = blk_reduce256(s, sred) / N;
  float var  = blk_reduce256(ss, sred) / N - mean * mean;
  float rstd = rsqrtf(var + 1e-5f);
  for (int e = threadIdx.x; e < N; e += 256)
    out[(size_t)b * N + e] = (zl[(size_t)b * N + e] - mean) * rstd * g[e] + be[e];
}

// ================================================================================
extern "C" void kernel_launch(void* const* d_in, const int* in_sizes, int n_in,
                              void* d_out, int out_size, void* d_ws, size_t ws_size,
                              hipStream_t stream) {
  const float* x0    = (const float*)d_in[0];
  const float* graph = (const float*)d_in[1];
  const float* w11 = (const float*)d_in[2];  const float* b11 = (const float*)d_in[3];
  const float* attw = (const float*)d_in[4]; const float* attb = (const float*)d_in[5];
  const float* th12 = (const float*)d_in[6]; const float* bb12 = (const float*)d_in[7];
  const float* w13 = (const float*)d_in[8];  const float* b13 = (const float*)d_in[9];
  const float* g1 = (const float*)d_in[10];  const float* be1 = (const float*)d_in[11];
  const float* w21 = (const float*)d_in[12]; const float* b21 = (const float*)d_in[13];
  const float* aw21 = (const float*)d_in[14];const float* ab21 = (const float*)d_in[15];
  const float* th22 = (const float*)d_in[16];const float* bb22 = (const float*)d_in[17];
  const float* w23 = (const float*)d_in[18]; const float* b23 = (const float*)d_in[19];
  const float* g2 = (const float*)d_in[20];  const float* be2 = (const float*)d_in[21];
  const float* wout = (const float*)d_in[22];const float* bout = (const float*)d_in[23];
  const float* g3 = (const float*)d_in[24];  const float* be3 = (const float*)d_in[25];
  const float* thi = (const float*)d_in[26]; const float* bbi = (const float*)d_in[27];
  const float* tho = (const float*)d_in[28]; const float* bbo = (const float*)d_in[29];
  const float* wlr = (const float*)d_in[30]; const float* blr = (const float*)d_in[31];
  const float* g4 = (const float*)d_in[32];  const float* be4 = (const float*)d_in[33];
  float* out = (float*)d_out;

  // ---- workspace partition (deterministic; ~575 MB) ----
  char* ws = (char*)d_ws;
  size_t off = 0;
  auto alloc = [&](size_t bytes) -> char* {
    char* p = ws + off; off += (bytes + 255) & ~(size_t)255; return p;
  };
  unsigned short* LkT  = (unsigned short*)alloc((size_t)Vn * 3 * Vn * 2);
  float* L    = (float*)alloc((size_t)Vn * Vn * 4);
  float* LL   = (float*)alloc((size_t)Vn * Vn * 4);
  unsigned short* LbfA = (unsigned short*)alloc((size_t)Vn * Vn * 2);
  unsigned short* LbfB = (unsigned short*)alloc((size_t)Vn * Vn * 2);
  float* dinv = (float*)alloc(Vn * 4);
  float* BUF1 = (float*)alloc((size_t)Bn * 32 * Tn * Vn * 4);      // t1 / s1 / s2
  float* BUF2 = (float*)alloc((size_t)Bn * 32 * T1 * Vn * 4);      // p1 / g2o
  unsigned short* AMIX = (unsigned short*)alloc((size_t)R1 * 3072 * 2);
  float* BUF5 = (float*)alloc((size_t)Bn * 64 * 44 * Vn * 4);      // r1 / r2
  float* XC   = (float*)alloc((size_t)Bn * 128 * Vn * 4);
  float* ZG   = (float*)alloc((size_t)Bn * 64 * Vn * 4);
  float* Z    = (float*)alloc((size_t)Bn * Vn * 64 * 4);
  float* RI   = (float*)alloc((size_t)Bn * Vn * 64 * 4);
  float* RO   = (float*)alloc((size_t)Bn * Vn * 64 * 4);
  float* ZL   = (float*)alloc((size_t)Bn * Vn * 64 * 4);
  (void)ws_size; (void)in_sizes; (void)n_in; (void)out_size;

  const int VV_BLK = (Vn * Vn) / 256;  // 4096

  // ---- graph Laplacian + Chebyshev (L@L via WMMA) ----
  k_degree   <<<Vn, 256, 0, stream>>>(graph, dinv);
  k_laplacian<<<VV_BLK, 256, 0, stream>>>(graph, dinv, L);
  k_cvtL     <<<VV_BLK, 256, 0, stream>>>(L, LbfA, LbfB);
  k_gemm_nt  <<<dim3(8, 16), 256, 0, stream>>>(LbfA, LbfB, LL, nullptr, nullptr,
                                               Vn, Vn, Vn, 1, 1, 0);
  k_buildLkT <<<VV_BLK, 256, 0, stream>>>(L, LL, LkT);

  // ---- stem: GLU temporal conv + attention pooler ----
  k_tconv1<<<dim3(4, Tn, Bn), 256, 0, stream>>>(x0, w11, b11, BUF1);
  k_pooler<<<dim3(Vn, Bn), 64, 0, stream>>>(BUF1, attw, attb, BUF2);

  // ---- sconv 1 (WMMA GEMM, fused bias+residual+ReLU) ----
  k_mix<<<(int)(((size_t)R1 * Vn + 255) / 256), 256, 0, stream>>>(BUF2, th12, AMIX, T1, R1);
  k_gemm_nt<<<dim3(8, R1 / 64), 256, 0, stream>>>(AMIX, LkT, BUF1, BUF2, bb12,
                                                  R1, Vn, 3072, T1, 32, 1);
  k_tconv_relu<<<dim3(4, 44, Bn), 256, 0, stream>>>(BUF1, w13, b13, BUF5, T1, 44);
  k_ln<<<dim3(44, Bn), 256, 0, stream>>>(BUF5, g1, be1, 64, 44);

  // ---- block 2 ----
  k_tconv_glu2<<<dim3(4, T2, Bn * 2), 256, 0, stream>>>(BUF5, w21, b21, aw21, ab21,
                                                        BUF2, 44, T2);
  k_mix<<<(int)(((size_t)R2 * Vn + 255) / 256), 256, 0, stream>>>(BUF2, th22, AMIX, T2, R2);
  k_gemm_nt<<<dim3(8, R2 / 64), 256, 0, stream>>>(AMIX, LkT, BUF1, BUF2, bb22,
                                                  R2, Vn, 3072, T2, 32, 1);
  k_tconv_relu<<<dim3(4, 40, Bn), 256, 0, stream>>>(BUF1, w23, b23, BUF5, T2, 40);
  k_ln<<<dim3(40, Bn), 256, 0, stream>>>(BUF5, g2, be2, 64, 40);

  // ---- output head ----
  k_tconv_out<<<dim3(4, 8, Bn), 256, 0, stream>>>(BUF5, wout, bout, XC);
  k_glu_out<<<dim3(4, 64, Bn), 256, 0, stream>>>(XC, BUF5, ZG);
  k_ln3<<<Bn, 256, 0, stream>>>(ZG, g3, be3, Z);
  k_res<<<dim3(Vn, Bn), 256, 0, stream>>>(x0, L, LL, thi, bbi, 0, RI);
  k_res<<<dim3(Vn, Bn), 256, 0, stream>>>(x0, L, LL, tho, bbo, 1, RO);
  k_linear<<<dim3(Vn, Bn), 64, 0, stream>>>(Z, RI, RO, wlr, blr, ZL);
  k_ln4<<<Bn, 256, 0, stream>>>(ZL, g4, be4, out);
}
